// RQVAE_54357106098815
// MI455X (gfx1250) — matrix operand order repaired and checked by reading, not compiled, not run on hardware
//
#include <hip/hip_runtime.h>
#include <stdint.h>

// ---------------------------------------------------------------------------
// RQ-VAE forward for MI455X (gfx1250): bf16 WMMA GEMMs with async global->LDS
// staging (ASYNCcnt) + f32 residual VQ.
// ---------------------------------------------------------------------------

typedef __attribute__((ext_vector_type(16))) __bf16 v16bf;
typedef __attribute__((ext_vector_type(8)))  float  v8f;

#define NROWS 65536
#define EDIM  128
#define NLEV  4
#define CBS   256
#define BETA  1.0e-3f

struct U32x8 { uint4 lo, hi; };

__device__ __forceinline__ v16bf frag_cast(uint4 lo, uint4 hi) {
    U32x8 t{lo, hi};
    return __builtin_bit_cast(v16bf, t);
}

__device__ __forceinline__ uint16_t f2bf(float f) {
    uint32_t u = __builtin_bit_cast(uint32_t, f);
    uint32_t r = u + 0x7FFFu + ((u >> 16) & 1u);   // round-to-nearest-even
    return (uint16_t)(r >> 16);
}

// gfx1250 async copy: global -> LDS, tracked with ASYNCcnt (ISA §10 / §15.18.3).
// DS address operand = low 32 bits of the generic LDS pointer (aperture rule).
__device__ __forceinline__ void async_b128(uint32_t lds_addr, const void* gaddr) {
    asm volatile("global_load_async_to_lds_b128 %0, %1, off"
                 :: "v"(lds_addr), "v"(gaddr)
                 : "memory");
}
__device__ __forceinline__ void wait_async0() {
    asm volatile("s_wait_asynccnt 0x0" ::: "memory");
}
__device__ __forceinline__ uint32_t lds_lo32(const void* p) {
    return (uint32_t)(uintptr_t)p;
}

// -------------------------------- prep kernels -----------------------------

__global__ void conv_f2b_kernel(const float* __restrict__ X,
                                uint16_t* __restrict__ Y, int total) {
    int i = blockIdx.x * blockDim.x + threadIdx.x;
    if (i < total) Y[i] = f2bf(X[i]);
}

// W (K x N, f32) -> Wt (N x K, bf16), output-indexed for coalesced stores
__global__ void conv_wt_kernel(const float* __restrict__ W,
                               uint16_t* __restrict__ Wt, int K, int N) {
    int i = blockIdx.x * blockDim.x + threadIdx.x;
    if (i >= K * N) return;
    int n = i / K, k = i - n * K;
    Wt[i] = f2bf(W[(size_t)k * N + n]);
}

// codebooks (4 x 256 x 128 f32) -> cbT (4 x 128 x 256) and norms (4 x 256)
__global__ void prep_cb_kernel(const float* __restrict__ cb,
                               float* __restrict__ cbT,
                               float* __restrict__ cbn) {
    int j = blockIdx.x * blockDim.x + threadIdx.x;
    if (j >= NLEV * CBS) return;
    int l = j >> 8, e = j & 255;
    const float* src = cb + (size_t)j * EDIM;
    float s = 0.f;
    for (int k = 0; k < EDIM; ++k) {
        float v = src[k];
        s += v * v;
        cbT[((size_t)l * EDIM + k) * CBS + e] = v;
    }
    cbn[j] = s;
}

__global__ void zero_loss_kernel(float* loss) {
    if (threadIdx.x == 0 && blockIdx.x == 0) *loss = 0.f;
}

// -------------------------------- GEMM kernel ------------------------------
// C[M,N] = act( A[M,K](bf16) @ Wt[N,K]^T(bf16) + bias[N] )
// 256 threads = 8 wave32 (4x2 wave grid); 256x128 block tile; each wave owns a
// 64x64 sub-tile = 4x4 WMMA accumulators. K-step 32, double-buffered LDS
// filled by async global->LDS copies.
template <bool RELU, bool OUT_BF16, bool OUT_F32>
__global__ __launch_bounds__(256) void gemm_bias_kernel(
    const uint16_t* __restrict__ A,   // M x K, bf16 row-major
    const uint16_t* __restrict__ Wt,  // N x K, bf16 row-major (pre-transposed)
    const float*    __restrict__ bias,
    uint16_t*       __restrict__ outb,
    float*          __restrict__ outf,
    int M, int K, int N)
{
    constexpr int LDT = 40;                 // padded LDS row stride (elements)
    __shared__ uint16_t lds_a[2][256 * LDT];
    __shared__ uint16_t lds_b[2][128 * LDT];

    const int tid  = threadIdx.x;
    const int bn   = blockIdx.x, bm = blockIdx.y;
    const int wave = tid >> 5,   lane = tid & 31;
    const int wr   = wave >> 1,  wc   = wave & 1;   // 4x2 wave grid
    const int half = lane >> 4,  m16  = lane & 15;

    const uint16_t* Ab = A  + (size_t)bm * 256 * K;
    const uint16_t* Bb = Wt + (size_t)bn * 128 * K;

    const uint32_t lab = lds_lo32(&lds_a[0][0]);
    const uint32_t lbb = lds_lo32(&lds_b[0][0]);

    v8f acc[4][4];
#pragma unroll
    for (int mt = 0; mt < 4; ++mt)
#pragma unroll
        for (int nt = 0; nt < 4; ++nt)
            acc[mt][nt] = (v8f){0.f, 0.f, 0.f, 0.f, 0.f, 0.f, 0.f, 0.f};

    const int nK = K >> 5;

    // async-stage one 32-wide K slice of A (256 rows) and B (128 rows)
    auto stage = [&](int buf, int kt) {
        const int k0 = kt << 5;
#pragma unroll
        for (int i = 0; i < 4; ++i) {               // A: 1024 16B chunks
            int idx = tid + 256 * i;
            int row = idx >> 2, c = idx & 3;
            async_b128(lab + (uint32_t)(buf * 256 * LDT + row * LDT + c * 8) * 2,
                       Ab + (size_t)row * K + k0 + c * 8);
        }
#pragma unroll
        for (int i = 0; i < 2; ++i) {               // B: 512 16B chunks
            int idx = tid + 256 * i;
            int row = idx >> 2, c = idx & 3;
            async_b128(lbb + (uint32_t)(buf * 128 * LDT + row * LDT + c * 8) * 2,
                       Bb + (size_t)row * K + k0 + c * 8);
        }
    };

    stage(0, 0);
    wait_async0();
    __syncthreads();

    for (int kt = 0; kt < nK; ++kt) {
        const int cur = kt & 1;
        if (kt + 1 < nK) stage(cur ^ 1, kt + 1);

        const uint16_t* la = lds_a[cur];
        const uint16_t* lb = lds_b[cur];

        v16bf bfv[4];
#pragma unroll
        for (int nt = 0; nt < 4; ++nt) {
            const uint16_t* p = lb + (wc * 64 + nt * 16 + m16) * LDT + half * 16;
            bfv[nt] = frag_cast(*(const uint4*)p, *(const uint4*)(p + 8));
        }
#pragma unroll
        for (int mt = 0; mt < 4; ++mt) {
            const uint16_t* p = la + (wr * 64 + mt * 16 + m16) * LDT + half * 16;
            v16bf af = frag_cast(*(const uint4*)p, *(const uint4*)(p + 8));
#pragma unroll
            for (int nt = 0; nt < 4; ++nt)
                acc[mt][nt] = __builtin_amdgcn_wmma_f32_16x16x32_bf16(
                    false, af, false, bfv[nt], (short)0, acc[mt][nt],
                    false, false);
        }

        wait_async0();        // this wave's async copies for next buffer done
        __syncthreads();      // everyone's copies visible / reads drained
    }

    // epilogue: bias (+ReLU), write bf16 and/or f32
#pragma unroll
    for (int mt = 0; mt < 4; ++mt) {
#pragma unroll
        for (int nt = 0; nt < 4; ++nt) {
            const int gr0 = bm * 256 + wr * 64 + mt * 16 + half * 8;
            const int gc  = bn * 128 + wc * 64 + nt * 16 + m16;
            const float bv = bias[gc];
#pragma unroll
            for (int r = 0; r < 8; ++r) {
                float v = acc[mt][nt][r] + bv;
                if (RELU) v = fmaxf(v, 0.f);
                size_t off = (size_t)(gr0 + r) * N + gc;
                if constexpr (OUT_BF16) outb[off] = f2bf(v);
                if constexpr (OUT_F32)  outf[off] = v;
            }
        }
    }
}

// ------------------------------ residual VQ --------------------------------
// 16 rows per block; thread = codebook entry; exact f32 argmin w/ tie-break.
__global__ __launch_bounds__(256) void vq_kernel(
    const float* __restrict__ z,        // N x 128
    const float* __restrict__ cb,       // 4 x 256 x 128
    const float* __restrict__ cbT,      // 4 x 128 x 256
    const float* __restrict__ cbn,      // 4 x 256
    float*       __restrict__ xq_f32,   // N x 128 (d_out)
    uint16_t*    __restrict__ xq_bf16,  // N x 128 (decoder input)
    int*         __restrict__ indices,  // N x 4
    float*       __restrict__ loss_accum)
{
    __shared__ float res[16 * EDIM];
    __shared__ float xq [16 * EDIM];
    __shared__ float smat[16 * CBS];
    __shared__ float red_s[256];
    __shared__ int   red_i[256];
    __shared__ int   bidx[16];

    const int tid = threadIdx.x;
    const size_t row0 = (size_t)blockIdx.x * 16;

#pragma unroll
    for (int i = 0; i < 8; ++i) {
        int e = tid + 256 * i;
        res[e] = z[row0 * EDIM + e];
        xq[e]  = 0.f;
    }
    __syncthreads();

    float lossAcc = 0.f;
    const int rr  = tid >> 4;   // row handled in reduce/update phases
    const int sub = tid & 15;

    for (int l = 0; l < NLEV; ++l) {
        // distances: score_j = ||cb_j||^2 - 2 * res . cb_j  (||res||^2 const)
        float acc[16];
#pragma unroll
        for (int r = 0; r < 16; ++r) acc[r] = 0.f;
        const float* ct = cbT + (size_t)l * EDIM * CBS;
        for (int k = 0; k < EDIM; ++k) {
            float c = ct[k * CBS + tid];
#pragma unroll
            for (int r = 0; r < 16; ++r) acc[r] += c * res[r * EDIM + k];
        }
        const float nrm = cbn[l * CBS + tid];
#pragma unroll
        for (int r = 0; r < 16; ++r) smat[r * CBS + tid] = nrm - 2.f * acc[r];
        __syncthreads();

        // argmin over 256 entries for row rr by its 16-thread group
        float best = 3.4e38f; int bi = 0x7fffffff;
        for (int i = 0; i < 16; ++i) {
            int j = sub + i * 16;
            float s = smat[rr * CBS + j];
            if (s < best || (s == best && j < bi)) { best = s; bi = j; }
        }
        red_s[tid] = best; red_i[tid] = bi;
        __syncthreads();
        for (int off = 8; off > 0; off >>= 1) {
            if (sub < off) {
                float so = red_s[tid + off]; int io = red_i[tid + off];
                if (so < red_s[tid] || (so == red_s[tid] && io < red_i[tid])) {
                    red_s[tid] = so; red_i[tid] = io;
                }
            }
            __syncthreads();
        }
        if (sub == 0) bidx[rr] = red_i[tid];
        __syncthreads();

        // update residual, accumulate x_q and loss
        const int q = bidx[rr];
        if (sub == 0) indices[(row0 + rr) * NLEV + l] = q;
        const float* cq = cb + ((size_t)l * CBS + q) * EDIM;
#pragma unroll
        for (int k8 = 0; k8 < 8; ++k8) {
            int k = sub * 8 + k8;
            float qv = cq[k];
            float rv = res[rr * EDIM + k];
            float d  = qv - rv;
            lossAcc += d * d;
            res[rr * EDIM + k] = rv - qv;   // res -= q_st (fwd value)
            xq [rr * EDIM + k] += qv;
        }
        __syncthreads();
    }

#pragma unroll
    for (int i = 0; i < 8; ++i) {
        int e = tid + 256 * i;
        float v = xq[e];
        xq_f32 [row0 * EDIM + e] = v;
        xq_bf16[row0 * EDIM + e] = f2bf(v);
    }

    // block loss reduction + single atomic
    red_s[tid] = lossAcc;
    __syncthreads();
    for (int off = 128; off > 0; off >>= 1) {
        if (tid < off) red_s[tid] += red_s[tid + off];
        __syncthreads();
    }
    if (tid == 0) {
        const float scale = (1.f + BETA) / ((float)NLEV * (float)NROWS * (float)EDIM);
        atomicAdd(loss_accum, red_s[0] * scale);
    }
}

// ------------------------------ host launcher ------------------------------

extern "C" void kernel_launch(void* const* d_in, const int* in_sizes, int n_in,
                              void* d_out, int out_size, void* d_ws, size_t ws_size,
                              hipStream_t stream) {
    (void)in_sizes; (void)n_in; (void)out_size; (void)ws_size;

    const float* x        = (const float*)d_in[0];
    const float* W[8]     = { (const float*)d_in[1],  (const float*)d_in[3],
                              (const float*)d_in[5],  (const float*)d_in[7],
                              (const float*)d_in[9],  (const float*)d_in[11],
                              (const float*)d_in[13], (const float*)d_in[15] };
    const float* Bv[8]    = { (const float*)d_in[2],  (const float*)d_in[4],
                              (const float*)d_in[6],  (const float*)d_in[8],
                              (const float*)d_in[10], (const float*)d_in[12],
                              (const float*)d_in[14], (const float*)d_in[16] };
    const float* cbooks   = (const float*)d_in[17];

    static const int Kd[8] = { 768, 2048, 1024, 512, 128, 512, 1024, 2048 };
    static const int Nd[8] = { 2048, 1024, 512, 128, 512, 1024, 2048, 768 };

    // d_out layout: out(65536x768) | rq_loss(1) | indices(65536x4) | x_q(65536x128)
    float* out_f  = (float*)d_out;
    float* loss_p = out_f + (size_t)NROWS * 768;
    int*   idx_p  = (int*)(loss_p + 1);
    float* xq_p   = (float*)(idx_p + (size_t)NROWS * NLEV);

    // workspace carve-out
    size_t off = 0;
    auto carve = [&](size_t bytes) {
        void* p = (char*)d_ws + off;
        off += (bytes + 255) & ~(size_t)255;
        return p;
    };
    uint16_t* act0 = (uint16_t*)carve((size_t)NROWS * 2048 * 2);
    uint16_t* act1 = (uint16_t*)carve((size_t)NROWS * 2048 * 2);
    uint16_t* Wt[8];
    for (int i = 0; i < 8; ++i)
        Wt[i] = (uint16_t*)carve((size_t)Kd[i] * Nd[i] * 2);
    float* zbuf = (float*)carve((size_t)NROWS * EDIM * 4);
    float* cbT  = (float*)carve((size_t)NLEV * EDIM * CBS * 4);
    float* cbn  = (float*)carve((size_t)NLEV * CBS * 4);

    // ---- prep ----
    {
        int total = NROWS * 768;
        conv_f2b_kernel<<<(total + 255) / 256, 256, 0, stream>>>(x, act0, total);
    }
    for (int i = 0; i < 8; ++i) {
        int total = Kd[i] * Nd[i];
        conv_wt_kernel<<<(total + 255) / 256, 256, 0, stream>>>(W[i], Wt[i], Kd[i], Nd[i]);
    }
    prep_cb_kernel<<<(NLEV * CBS + 255) / 256, 256, 0, stream>>>(cbooks, cbT, cbn);
    zero_loss_kernel<<<1, 1, 0, stream>>>(loss_p);

    const int MB = NROWS / 256;
    auto grid = [&](int li) { return dim3(Nd[li] / 128, MB, 1); };

    // ---- encoder ----
    gemm_bias_kernel<true, true, false><<<grid(0), 256, 0, stream>>>(
        act0, Wt[0], Bv[0], act1, nullptr, NROWS, Kd[0], Nd[0]);
    gemm_bias_kernel<true, true, false><<<grid(1), 256, 0, stream>>>(
        act1, Wt[1], Bv[1], act0, nullptr, NROWS, Kd[1], Nd[1]);
    gemm_bias_kernel<true, true, false><<<grid(2), 256, 0, stream>>>(
        act0, Wt[2], Bv[2], act1, nullptr, NROWS, Kd[2], Nd[2]);
    gemm_bias_kernel<false, false, true><<<grid(3), 256, 0, stream>>>(
        act1, Wt[3], Bv[3], nullptr, zbuf, NROWS, Kd[3], Nd[3]);

    // ---- residual VQ ----  (xq bf16 -> act0 for decoder input)
    vq_kernel<<<NROWS / 16, 256, 0, stream>>>(
        zbuf, cbooks, cbT, cbn, xq_p, act0, idx_p, loss_p);

    // ---- decoder ----
    gemm_bias_kernel<true, true, false><<<grid(4), 256, 0, stream>>>(
        act0, Wt[4], Bv[4], act1, nullptr, NROWS, Kd[4], Nd[4]);
    gemm_bias_kernel<true, true, false><<<grid(5), 256, 0, stream>>>(
        act1, Wt[5], Bv[5], act0, nullptr, NROWS, Kd[5], Nd[5]);
    gemm_bias_kernel<true, true, false><<<grid(6), 256, 0, stream>>>(
        act0, Wt[6], Bv[6], act1, nullptr, NROWS, Kd[6], Nd[6]);
    gemm_bias_kernel<false, false, true><<<grid(7), 256, 0, stream>>>(
        act1, Wt[7], Bv[7], nullptr, out_f, NROWS, Kd[7], Nd[7]);
}
